// ConcatAttention_19774029431577
// MI455X (gfx1250) — matrix operand order repaired and checked by reading, not compiled, not run on hardware
//
#include <hip/hip_runtime.h>
#include <hip/hip_bf16.h>

typedef _Float16 v4h  __attribute__((ext_vector_type(4)));
typedef _Float16 v8h  __attribute__((ext_vector_type(8)));
typedef _Float16 v16h __attribute__((ext_vector_type(16)));
typedef float    v8f  __attribute__((ext_vector_type(8)));

#define B_SZ   32
#define S_SZ   2048
#define ENC_D  1024
#define HID_D  1024
#define DEC_D  1024
#define M_TILE 64
#define MASK_VAL -50000.0f

__device__ __forceinline__ float fast_tanh(float x) {
#if __has_builtin(__builtin_amdgcn_tanhf)
    return __builtin_amdgcn_tanhf(x);
#elif __has_builtin(__builtin_amdgcn_tanh_f32)
    return __builtin_amdgcn_tanh_f32(x);
#else
    return tanhf(x);
#endif
}

// ---------------------------------------------------------------------------
// Kernel 1: convert W_enc (first 1024 cols of W[h][2048]) to f16 [h][k]
// ---------------------------------------------------------------------------
__global__ __launch_bounds__(256) void convert_w_kernel(
    const float* __restrict__ W, _Float16* __restrict__ w16)
{
    int idx = blockIdx.x * 256 + threadIdx.x;   // float4 index: 1024*1024/4
    int h   = idx >> 8;                          // 256 float4 per row
    int k4  = idx & 255;
    float4 f = ((const float4*)(W + (size_t)h * 2048))[k4];
    v4h o = { (_Float16)f.x, (_Float16)f.y, (_Float16)f.z, (_Float16)f.w };
    *(v4h*)(w16 + (size_t)h * 1024 + k4 * 4) = o;
}

// ---------------------------------------------------------------------------
// Kernel 2: dec_proj[b][h] = sum_d dec[b][d] * W[h][1024+d]   (f32)
// ---------------------------------------------------------------------------
__global__ __launch_bounds__(256) void dec_proj_kernel(
    const float* __restrict__ W, const float* __restrict__ dec,
    float* __restrict__ dproj)
{
    int h = blockIdx.x * 256 + threadIdx.x;
    int b = blockIdx.y;
    const float* wr = W + (size_t)h * 2048 + 1024;
    const float* dr = dec + (size_t)b * DEC_D;
    float acc = 0.0f;
    #pragma unroll 4
    for (int d = 0; d < DEC_D; d += 4) {
        float4 w4 = *(const float4*)(wr + d);
        float4 d4 = *(const float4*)(dr + d);
        acc = fmaf(w4.x, d4.x, acc);
        acc = fmaf(w4.y, d4.y, acc);
        acc = fmaf(w4.z, d4.z, acc);
        acc = fmaf(w4.w, d4.w, acc);
    }
    dproj[(size_t)b * HID_D + h] = acc;
}

// ---------------------------------------------------------------------------
// Kernel 3: fused scores. Per workgroup: one (b, 64-row s-tile).
// 8 waves = 4 m-subtiles x 2 n-halves. f16 WMMA, f32 accum,
// fused tanh + dot-with-v epilogue. Writes logits[b][s] (pre-mask).
// 132 KB LDS A-tile -> each workgroup reads W16 (2MB, L2-resident) once
// per 64 rows instead of once per 16 rows: 4x less L2 traffic.
// ---------------------------------------------------------------------------
__global__ __launch_bounds__(256) void attn_scores_kernel(
    const float*    __restrict__ enc,
    const _Float16* __restrict__ w16,
    const float*    __restrict__ dproj,
    const float*    __restrict__ v,
    float*          __restrict__ logits)
{
    __shared__ _Float16 tileA[M_TILE][1032]; // +8 halfs pad: stride 2064B=129*16B
    __shared__ float    logit_s[M_TILE];

    const int s0   = blockIdx.x * M_TILE;
    const int b    = blockIdx.y;
    const int tid  = threadIdx.x;
    const int wave = tid >> 5;
    const int lane = tid & 31;
    const int ln16 = lane & 15;
    const bool hiHalf = (lane >= 16);
    const int msub  = wave & 3;   // which 16-row slice of the 64-row tile
    const int nhalf = wave >> 2;  // which 512-wide half of H

    if (tid < M_TILE) logit_s[tid] = 0.0f;

    // Stage enc tile (64 x 1024) f32 -> f16 into LDS (coalesced float4 reads)
    const float* encB = enc + ((size_t)b * S_SZ + s0) * ENC_D;
    #pragma unroll
    for (int i = 0; i < (M_TILE * ENC_D / 4) / 256; ++i) {
        int gidx = i * 256 + tid;              // 16384 float4s total
        float4 f = ((const float4*)encB)[gidx];
        int row = gidx >> 8;
        int col = (gidx & 255) * 4;
        v4h h4 = { (_Float16)f.x, (_Float16)f.y, (_Float16)f.z, (_Float16)f.w };
        *(v4h*)&tileA[row][col] = h4;
    }
    __syncthreads();

    float lp[8];
    #pragma unroll
    for (int r = 0; r < 8; ++r) lp[r] = 0.0f;

    // A fragment addressing (ISA 7.12.2, 16-bit A 16x32):
    //   lanes 0-15 : M=lane,   K = {0..7, 16..23}
    //   lanes16-31 : M=lane-16,K = {8..15,24..31}
    const int aRow  = msub * 16 + ln16;
    const int aKoff = hiHalf ? 8 : 0;
    // B fragment (32x16, column per lane): lanes 0-15 K={0..7,16..23}, 16-31 K={8..15,24..31}
    const int bKoff = hiHalf ? 8 : 0;

    for (int ntc = 0; ntc < 8; ++ntc) {
        v8f c[4];
        #pragma unroll
        for (int j = 0; j < 4; ++j)
            #pragma unroll
            for (int r = 0; r < 8; ++r) c[j][r] = 0.0f;

        for (int k0 = 0; k0 < ENC_D; k0 += 32) {
            v8h alo = *(const v8h*)&tileA[aRow][k0 + aKoff];
            v8h ahi = *(const v8h*)&tileA[aRow][k0 + aKoff + 16];
            v16h a = __builtin_shufflevector(alo, ahi,
                0,1,2,3,4,5,6,7,8,9,10,11,12,13,14,15);
            #pragma unroll
            for (int j = 0; j < 4; ++j) {
                int n = nhalf * 512 + (ntc * 4 + j) * 16 + ln16;
                const _Float16* wp = w16 + (size_t)n * 1024 + k0 + bKoff;
                v8h blo = *(const v8h*)wp;
                v8h bhi = *(const v8h*)(wp + 16);
                v16h bb = __builtin_shufflevector(blo, bhi,
                    0,1,2,3,4,5,6,7,8,9,10,11,12,13,14,15);
                c[j] = __builtin_amdgcn_wmma_f32_16x16x32_f16(
                    false, a, false, bb, (short)0, c[j], false, false);
            }
        }

        // Epilogue: hidden = tanh(acc + dec_proj[b][n]); lp += hidden * v[n]
        // C layout: lane<16: (M=r, N=ln16); lane>=16: (M=r+8, N=ln16)
        #pragma unroll
        for (int j = 0; j < 4; ++j) {
            int n = nhalf * 512 + (ntc * 4 + j) * 16 + ln16;
            float dp = dproj[(size_t)b * HID_D + n];
            float vv = v[n];
            #pragma unroll
            for (int r = 0; r < 8; ++r)
                lp[r] = fmaf(fast_tanh(c[j][r] + dp), vv, lp[r]);
        }
    }

    // Reduce over the 16 N-lanes within each half-wave
    #pragma unroll
    for (int m = 1; m < 16; m <<= 1) {
        #pragma unroll
        for (int r = 0; r < 8; ++r) lp[r] += __shfl_xor(lp[r], m, 32);
    }
    if (ln16 == 0) {
        int mbase = msub * 16 + (hiHalf ? 8 : 0);
        #pragma unroll
        for (int r = 0; r < 8; ++r) atomicAdd(&logit_s[mbase + r], lp[r]);
    }
    __syncthreads();
    if (tid < M_TILE)
        logits[(size_t)b * S_SZ + s0 + tid] = logit_s[tid];
}

// ---------------------------------------------------------------------------
// Kernel 4: masked softmax over S per batch -> attn_weight (d_out slot 2)
// ---------------------------------------------------------------------------
__global__ __launch_bounds__(256) void softmax_kernel(
    const float* __restrict__ logits, const int* __restrict__ seqlen,
    float* __restrict__ attn)
{
    __shared__ float red[8];
    const int b = blockIdx.x;
    const int t = threadIdx.x;
    const int len = seqlen[b];

    float vals[8];
    float mx = -3.0e38f;
    #pragma unroll
    for (int i = 0; i < 8; ++i) {
        int s = i * 256 + t;
        float x = logits[(size_t)b * S_SZ + s];
        x = (s < len) ? x : MASK_VAL;
        vals[i] = x;
        mx = fmaxf(mx, x);
    }
    #pragma unroll
    for (int m = 1; m < 32; m <<= 1) mx = fmaxf(mx, __shfl_xor(mx, m, 32));
    if ((t & 31) == 0) red[t >> 5] = mx;
    __syncthreads();
    mx = red[0];
    #pragma unroll
    for (int i = 1; i < 8; ++i) mx = fmaxf(mx, red[i]);
    __syncthreads();

    float sum = 0.0f;
    #pragma unroll
    for (int i = 0; i < 8; ++i) {
        float e = __expf(vals[i] - mx);
        vals[i] = e;
        sum += e;
    }
    #pragma unroll
    for (int m = 1; m < 32; m <<= 1) sum += __shfl_xor(sum, m, 32);
    if ((t & 31) == 0) red[t >> 5] = sum;
    __syncthreads();
    sum = 0.0f;
    #pragma unroll
    for (int i = 0; i < 8; ++i) sum += red[i];
    float inv = 1.0f / sum;

    #pragma unroll
    for (int i = 0; i < 8; ++i)
        attn[(size_t)b * S_SZ + i * 256 + t] = vals[i] * inv;
}

// ---------------------------------------------------------------------------
// Kernel 5: out[b][e] = sum_s attn[b][s] * enc[b][s][e]   (bandwidth-bound)
// ---------------------------------------------------------------------------
__global__ __launch_bounds__(256) void context_kernel(
    const float* __restrict__ enc, const float* __restrict__ attn,
    float* __restrict__ out)
{
    __shared__ float aS[256];
    const int b = blockIdx.y;
    const int e = blockIdx.x * 256 + threadIdx.x;
    const float* encB = enc + (size_t)b * S_SZ * ENC_D;

    float acc = 0.0f;
    for (int sc = 0; sc < S_SZ; sc += 256) {
        __syncthreads();
        aS[threadIdx.x] = attn[(size_t)b * S_SZ + sc + threadIdx.x];
        __syncthreads();
        #pragma unroll 8
        for (int ss = 0; ss < 256; ++ss)
            acc = fmaf(aS[ss], encB[(size_t)(sc + ss) * ENC_D + e], acc);
    }
    out[(size_t)b * ENC_D + e] = acc;
}

// ---------------------------------------------------------------------------
extern "C" void kernel_launch(void* const* d_in, const int* in_sizes, int n_in,
                              void* d_out, int out_size, void* d_ws, size_t ws_size,
                              hipStream_t stream)
{
    const float* enc    = (const float*)d_in[0];   // (32, 2048, 1024) f32
    const int*   seqlen = (const int*)  d_in[1];   // (32,) int
    const float* dec    = (const float*)d_in[2];   // (32, 1024) f32
    const float* W      = (const float*)d_in[3];   // (1024, 2048) f32
    const float* v      = (const float*)d_in[4];   // (1024,) f32

    float* out  = (float*)d_out;                   // (32, 1024)
    float* attn = out + (size_t)B_SZ * ENC_D;      // (32, 2048)

    char* ws = (char*)d_ws;
    _Float16* w16   = (_Float16*)ws;                                   // 2 MB
    float*    dproj = (float*)(ws + (size_t)HID_D * ENC_D * 2);        // 128 KB
    float*    logit = (float*)(ws + (size_t)HID_D * ENC_D * 2
                                  + (size_t)B_SZ * HID_D * 4);         // 256 KB

    convert_w_kernel <<<dim3(1024),      256, 0, stream>>>(W, w16);
    dec_proj_kernel  <<<dim3(4, B_SZ),   256, 0, stream>>>(W, dec, dproj);
    attn_scores_kernel<<<dim3(S_SZ / M_TILE, B_SZ), 256, 0, stream>>>(
        enc, w16, dproj, v, logit);
    softmax_kernel   <<<dim3(B_SZ),      256, 0, stream>>>(logit, seqlen, attn);
    context_kernel   <<<dim3(ENC_D / 256, B_SZ), 256, 0, stream>>>(enc, attn, out);
}
